// RelPosSelfAttention_72748156060259
// MI455X (gfx1250) — matrix operand order repaired and checked by reading, not compiled, not run on hardware
//
#include <hip/hip_runtime.h>
#include <hip/hip_bf16.h>
#include <math.h>

// Problem constants (fixed by the reference)
#define DM   1024
#define NH   16
#define DH   64
#define LSEQ 2048
#define NB   2
#define D3   3072
#define MTOT (NB * LSEQ) // 4096
#define MAXREL 512

typedef __attribute__((ext_vector_type(16))) __bf16 v16bf;
typedef __attribute__((ext_vector_type(8)))  float  v8f;

union FragAB { v16bf v; uint4 q[2]; unsigned short us[16]; };
union FragC  { v8f v; float f[8]; };

__device__ __forceinline__ unsigned short f2bf(float f) {
  unsigned u = __float_as_uint(f);
  unsigned r = u + 0x7FFFu + ((u >> 16) & 1u); // round-to-nearest-even
  return (unsigned short)(r >> 16);
}
__device__ __forceinline__ int iclamp(int v, int lo, int hi) {
  return v < lo ? lo : (v > hi ? hi : v);
}

// ---------------- conversion kernels ----------------
__global__ void rpsa_cvt_bf16(const float* __restrict__ in,
                              unsigned short* __restrict__ out, int n) {
  int i = blockIdx.x * blockDim.x + threadIdx.x;
  int stride = gridDim.x * blockDim.x;
  for (; i < n; i += stride) out[i] = f2bf(in[i]);
}

// in: [K,N] f32 row-major  ->  out: [N,K] bf16 (transposed)
__global__ void rpsa_cvt_bf16_T(const float* __restrict__ in,
                                unsigned short* __restrict__ out, int K, int N) {
  int i = blockIdx.x * blockDim.x + threadIdx.x;
  int stride = gridDim.x * blockDim.x;
  int total = K * N;
  for (; i < total; i += stride) {
    int k = i / N, n = i - k * N;
    out[(size_t)n * K + k] = f2bf(in[i]);
  }
}

// ---------------- WMMA GEMM: C[M,N] = A[M,K] * Bt[N,K]^T + bias ----------------
// A, Bt bf16; acc f32. Block tile 128(M) x 256(N), 256 threads = 8 waves (2x4),
// each wave computes a 64x64 tile via 4x4 fragments (16 WMMAs / K-step).
// K-step 32, register double-buffering of global->LDS.
template <bool OUTBF16>
__global__ __launch_bounds__(256)
void rpsa_gemm_wmma(const unsigned short* __restrict__ A,
                    const unsigned short* __restrict__ Bt,
                    const float* __restrict__ bias,
                    void* __restrict__ Cout, int M, int N, int K) {
  __shared__ __align__(16) unsigned short Asub[128 * 48]; // stride 48 bf16 (96B)
  __shared__ __align__(16) unsigned short Bsub[256 * 48];

  const int tid = threadIdx.x;
  const int lane = tid & 31, wid = tid >> 5;
  const int g = lane >> 4, ln = lane & 15;
  const int wm = wid & 1, wn = wid >> 1; // 2 x 4 wave grid
  const int m0 = blockIdx.y * 128;
  const int n0 = blockIdx.x * 256;

  FragC acc[4][4];
#pragma unroll
  for (int i = 0; i < 4; i++)
#pragma unroll
    for (int j = 0; j < 4; j++)
#pragma unroll
      for (int r = 0; r < 8; r++) acc[i][j].f[r] = 0.0f;

  // global->register staging layout
  const int arow = tid >> 1;            // 0..127
  const int ahalf = (tid & 1) * 16;     // 0 or 16 bf16 elems
  const int brow = tid;                 // 0..255, loads full 32-k row

  const unsigned short* aptr = A + (size_t)(m0 + arow) * K + ahalf;
  const unsigned short* bptr = Bt + (size_t)(n0 + brow) * K;

  uint4 pa0, pa1, pb0, pb1, pb2, pb3;
  pa0 = *(const uint4*)(aptr);
  pa1 = *(const uint4*)(aptr + 8);
  pb0 = *(const uint4*)(bptr);
  pb1 = *(const uint4*)(bptr + 8);
  pb2 = *(const uint4*)(bptr + 16);
  pb3 = *(const uint4*)(bptr + 24);

  for (int kk = 0; kk < K; kk += 32) {
    __syncthreads(); // previous tile fully consumed
    *(uint4*)&Asub[arow * 48 + ahalf]     = pa0;
    *(uint4*)&Asub[arow * 48 + ahalf + 8] = pa1;
    *(uint4*)&Bsub[brow * 48 + 0]  = pb0;
    *(uint4*)&Bsub[brow * 48 + 8]  = pb1;
    *(uint4*)&Bsub[brow * 48 + 16] = pb2;
    *(uint4*)&Bsub[brow * 48 + 24] = pb3;
    __syncthreads();

    if (kk + 32 < K) { // prefetch next K-slice into registers during the math
      pa0 = *(const uint4*)(aptr + kk + 32);
      pa1 = *(const uint4*)(aptr + kk + 40);
      pb0 = *(const uint4*)(bptr + kk + 32);
      pb1 = *(const uint4*)(bptr + kk + 40);
      pb2 = *(const uint4*)(bptr + kk + 48);
      pb3 = *(const uint4*)(bptr + kk + 56);
    }

    FragAB aF[4];
#pragma unroll
    for (int fm = 0; fm < 4; fm++) { // A frag: row = m; e0..7 -> k=g*8+e, e8..15 -> 16+g*8+(e-8)
      int m = wm * 64 + fm * 16 + ln;
      aF[fm].q[0] = *(const uint4*)&Asub[m * 48 + g * 8];
      aF[fm].q[1] = *(const uint4*)&Asub[m * 48 + 16 + g * 8];
    }
#pragma unroll
    for (int fn = 0; fn < 4; fn++) { // B frag: col = n; e0..7 -> k=g*8+e, e8..15 -> 16+g*8+(e-8)
      int n = wn * 64 + fn * 16 + ln;
      FragAB bF;
      bF.q[0] = *(const uint4*)&Bsub[n * 48 + g * 8];
      bF.q[1] = *(const uint4*)&Bsub[n * 48 + 16 + g * 8];
#pragma unroll
      for (int fm = 0; fm < 4; fm++)
        acc[fm][fn].v = __builtin_amdgcn_wmma_f32_16x16x32_bf16(
            false, aF[fm].v, false, bF.v, (short)0, acc[fm][fn].v, false, false);
    }
  }

#pragma unroll
  for (int fm = 0; fm < 4; fm++) {
#pragma unroll
    for (int fn = 0; fn < 4; fn++) {
      int n = n0 + wn * 64 + fn * 16 + ln;
      float bv = bias[n];
#pragma unroll
      for (int r = 0; r < 8; r++) { // C: m = r + 8*g, n = ln
        int row = m0 + wm * 64 + fm * 16 + r + 8 * g;
        float val = acc[fm][fn].f[r] + bv;
        if (OUTBF16)
          ((unsigned short*)Cout)[(size_t)row * N + n] = f2bf(val);
        else
          ((float*)Cout)[(size_t)row * N + n] = val;
      }
    }
  }
}

// ---------------- flash attention with rel-pos bias ----------------
// grid: (L/128, H, B); block 256 = 8 waves; wave owns 16 q rows.
__global__ __launch_bounds__(256)
void rpsa_attn_wmma(const unsigned short* __restrict__ qkv, // [B*L, 3072] bf16
                    const unsigned char* __restrict__ kpm,  // [B, L] bool
                    const float* __restrict__ rel_table,    // [1025, NH] f32
                    unsigned short* __restrict__ outb) {    // [B*L, 1024] bf16
  __shared__ __align__(16) unsigned short Kt[32 * 80];   // [key][d], stride 80
  __shared__ __align__(16) unsigned short Vt[64 * 48];   // [d][key], stride 48 (transposed)
  __shared__ __align__(16) unsigned short Pst[8][16 * 32]; // per-wave P staging
  __shared__ float relcol[1032];

  const int tid = threadIdx.x;
  const int lane = tid & 31, wid = tid >> 5;
  const int g = lane >> 4, ln = lane & 15;
  const int h = blockIdx.y, b = blockIdx.z;
  const int q0 = blockIdx.x * 128 + wid * 16;
  const float NEGINF = -__builtin_inff();

  for (int i = tid; i < 2 * MAXREL + 1; i += 256)
    relcol[i] = rel_table[i * NH + h];

  // Q fragments (held in registers for the whole kernel)
  const unsigned short* qrow = qkv + ((size_t)(b * LSEQ) + q0 + ln) * D3 + h * DH;
  FragAB aQ[2];
#pragma unroll
  for (int f = 0; f < 2; f++) {
    aQ[f].q[0] = *(const uint4*)(qrow + f * 32 + g * 8);
    aQ[f].q[1] = *(const uint4*)(qrow + f * 32 + 16 + g * 8);
  }

  float mi[8], li[8];
  FragC acc[4];
#pragma unroll
  for (int r = 0; r < 8; r++) { mi[r] = NEGINF; li[r] = 0.0f; }
#pragma unroll
  for (int nt = 0; nt < 4; nt++)
#pragma unroll
    for (int r = 0; r < 8; r++) acc[nt].f[r] = 0.0f;

  const int ldrow = tid >> 3;        // 0..31 (key within tile)
  const int ldcol = (tid & 7) * 8;   // 0..56 (d)

  for (int key0 = 0; key0 < LSEQ; key0 += 32) {
    __syncthreads();
    { // K tile, natural [key][d]
      const unsigned short* src =
          qkv + ((size_t)(b * LSEQ) + key0 + ldrow) * D3 + DM + h * DH + ldcol;
      *(uint4*)&Kt[ldrow * 80 + ldcol] = *(const uint4*)src;
    }
    { // V tile, transposed into [d][key]
      const unsigned short* src =
          qkv + ((size_t)(b * LSEQ) + key0 + ldrow) * D3 + 2 * DM + h * DH + ldcol;
      uint4 t16 = *(const uint4*)src;
      unsigned short tmp[8];
      *(uint4*)tmp = t16;
#pragma unroll
      for (int j = 0; j < 8; j++) Vt[(ldcol + j) * 48 + ldrow] = tmp[j];
    }
    __syncthreads();

    // scores S[s] = Q @ K^T for key halves s=0,1 (16 keys each)
    FragC S[2];
#pragma unroll
    for (int s = 0; s < 2; s++) {
      FragAB bk0, bk1; // B frag: n = key, k = d; e0..7 -> d = dbase + g*8 + e, e8..15 -> dbase + 16 + g*8 + (e-8)
      int key = s * 16 + ln;
      bk0.q[0] = *(const uint4*)&Kt[key * 80 + g * 8];
      bk0.q[1] = *(const uint4*)&Kt[key * 80 + 16 + g * 8];
      bk1.q[0] = *(const uint4*)&Kt[key * 80 + 32 + g * 8];
      bk1.q[1] = *(const uint4*)&Kt[key * 80 + 32 + 16 + g * 8];
      FragC z;
#pragma unroll
      for (int r = 0; r < 8; r++) z.f[r] = 0.0f;
      z.v = __builtin_amdgcn_wmma_f32_16x16x32_bf16(false, aQ[0].v, false, bk0.v,
                                                    (short)0, z.v, false, false);
      z.v = __builtin_amdgcn_wmma_f32_16x16x32_bf16(false, aQ[1].v, false, bk1.v,
                                                    (short)0, z.v, false, false);
      S[s] = z;
    }

    bool msk0 = kpm[(size_t)b * LSEQ + key0 + ln] != 0;
    bool msk1 = kpm[(size_t)b * LSEQ + key0 + 16 + ln] != 0;

    float p0[8], p1[8], alpha[8];
#pragma unroll
    for (int r = 0; r < 8; r++) {
      int qpos = q0 + r + 8 * g;
      int k0 = key0 + ln, k1 = key0 + 16 + ln;
      float s0 = S[0].f[r] * 0.125f + relcol[iclamp(k0 - qpos, -MAXREL, MAXREL) + MAXREL];
      float s1 = S[1].f[r] * 0.125f + relcol[iclamp(k1 - qpos, -MAXREL, MAXREL) + MAXREL];
      if (msk0) s0 = NEGINF;
      if (msk1) s1 = NEGINF;
      float mx = fmaxf(s0, s1);
#pragma unroll
      for (int off = 1; off < 16; off <<= 1) mx = fmaxf(mx, __shfl_xor(mx, off, 32));
      float mnew = fmaxf(mi[r], mx);
      float a = (mnew == NEGINF) ? 1.0f : __expf(mi[r] - mnew);
      float e0 = (mnew == NEGINF) ? 0.0f : __expf(s0 - mnew);
      float e1 = (mnew == NEGINF) ? 0.0f : __expf(s1 - mnew);
      float rs = e0 + e1;
#pragma unroll
      for (int off = 1; off < 16; off <<= 1) rs += __shfl_xor(rs, off, 32);
      li[r] = li[r] * a + rs;
      mi[r] = mnew;
      alpha[r] = a;
      p0[r] = e0;
      p1[r] = e1;
    }

    // re-layout P: C-frag -> per-wave LDS -> A-frag (wave-private, DS in-order)
    unsigned short* pw = Pst[wid];
#pragma unroll
    for (int r = 0; r < 8; r++) {
      int m = r + 8 * g;
      pw[m * 32 + ln]      = f2bf(p0[r]);
      pw[m * 32 + 16 + ln] = f2bf(p1[r]);
    }
#pragma unroll
    for (int nt = 0; nt < 4; nt++)
#pragma unroll
      for (int r = 0; r < 8; r++) acc[nt].f[r] *= alpha[r];

    FragAB aP;
    aP.q[0] = *(const uint4*)&pw[ln * 32 + g * 8];
    aP.q[1] = *(const uint4*)&pw[ln * 32 + 16 + g * 8];

#pragma unroll
    for (int nt = 0; nt < 4; nt++) { // B frag from transposed V: n = d, k = key
      FragAB bV;
      int d = nt * 16 + ln;
      bV.q[0] = *(const uint4*)&Vt[d * 48 + g * 8];
      bV.q[1] = *(const uint4*)&Vt[d * 48 + 16 + g * 8];
      acc[nt].v = __builtin_amdgcn_wmma_f32_16x16x32_bf16(false, aP.v, false, bV.v,
                                                          (short)0, acc[nt].v, false, false);
    }
  }

#pragma unroll
  for (int r = 0; r < 8; r++) {
    float inv = li[r] > 0.0f ? 1.0f / li[r] : 0.0f;
    size_t row = (size_t)(b * LSEQ) + q0 + r + 8 * g;
#pragma unroll
    for (int nt = 0; nt < 4; nt++)
      outb[row * DM + h * DH + nt * 16 + ln] = f2bf(acc[nt].f[r] * inv);
  }
}

// ---------------- launch ----------------
extern "C" void kernel_launch(void* const* d_in, const int* in_sizes, int n_in,
                              void* d_out, int out_size, void* d_ws, size_t ws_size,
                              hipStream_t stream) {
  const float*         x    = (const float*)d_in[0];
  const unsigned char* kpm  = (const unsigned char*)d_in[1];
  const float*         Wqkv = (const float*)d_in[2];
  const float*         bqkv = (const float*)d_in[3];
  const float*         Wout = (const float*)d_in[4];
  const float*         bout = (const float*)d_in[5];
  const float*         rel  = (const float*)d_in[6];
  float* out = (float*)d_out;

  unsigned short* xb    = (unsigned short*)d_ws;                    // [4096,1024]
  unsigned short* wqb   = xb    + (size_t)MTOT * DM;                // [3072,1024] (W^T)
  unsigned short* wob   = wqb   + (size_t)D3 * DM;                  // [1024,1024] (W^T)
  unsigned short* qkvb  = wob   + (size_t)DM * DM;                  // [4096,3072]
  unsigned short* attnb = qkvb  + (size_t)MTOT * D3;                // [4096,1024]

  rpsa_cvt_bf16<<<2048, 256, 0, stream>>>(x, xb, MTOT * DM);
  rpsa_cvt_bf16_T<<<2048, 256, 0, stream>>>(Wqkv, wqb, DM, D3);
  rpsa_cvt_bf16_T<<<2048, 256, 0, stream>>>(Wout, wob, DM, DM);

  // QKV projection: [4096,1024] x [1024,3072] + bqkv -> bf16
  rpsa_gemm_wmma<true><<<dim3(D3 / 256, MTOT / 128), 256, 0, stream>>>(
      xb, wqb, bqkv, (void*)qkvb, MTOT, D3, DM);

  // Attention
  rpsa_attn_wmma<<<dim3(LSEQ / 128, NH, NB), 256, 0, stream>>>(qkvb, kpm, rel, attnb);

  // Output projection: [4096,1024] x [1024,1024] + bout -> f32
  rpsa_gemm_wmma<false><<<dim3(DM / 256, MTOT / 128), 256, 0, stream>>>(
      attnb, wob, bout, (void*)out, MTOT, DM, DM);
}
